// MultiHeadAttention_9002251452758
// MI455X (gfx1250) — matrix operand order repaired and checked by reading, not compiled, not run on hardware
//
#include <hip/hip_runtime.h>
#include <hip/hip_bf16.h>

// ---------------- problem constants ----------------
#define NB   2
#define LQ   2048
#define SK   2048
#define EM   1024
#define HH   16
#define DH   64
#define NT   (NB * LQ)          // 4096 total query tokens

#define AV_CHUNK 512            // S-chunk staged in LDS per TDM op
#define AV_CHUNK_BYTES (DH * AV_CHUNK * 2)   // 64 KB

typedef __attribute__((ext_vector_type(16))) _Float16 v16h;
typedef __attribute__((ext_vector_type(8)))  _Float16 v8h;
typedef __attribute__((ext_vector_type(8)))  float    v8f;
typedef __attribute__((ext_vector_type(4)))  unsigned int v4u;
typedef __attribute__((ext_vector_type(8)))  int v8i;
typedef __attribute__((ext_vector_type(4)))  int v4i;

// ---------------- WMMA fragment helpers (wave32, 16x16x32 f16 -> f32) -------
// A 16x32 (MxK) f16: lane-half 0 holds K in {0..7, 16..23}, half 1 {8..15, 24..31}
__device__ __forceinline__ v16h load_a_f16(const _Float16* __restrict__ base, int ld) {
    int lane = threadIdx.x & 31;
    int half = lane >> 4;
    int m    = lane & 15;
    const _Float16* p = base + (size_t)m * ld + 8 * half;
    v8h lo = *(const v8h*)p;          // K = 8*half + 0..7
    v8h hi = *(const v8h*)(p + 16);   // K = 16 + 8*half + 0..7
    v16h r;
#pragma unroll
    for (int i = 0; i < 8; ++i) { r[i] = lo[i]; r[i + 8] = hi[i]; }
    return r;
}

// Same A fragment but source is f32 (attn probabilities), convert per element.
__device__ __forceinline__ v16h load_a_f32cvt(const float* __restrict__ base, int ld) {
    int lane = threadIdx.x & 31;
    int half = lane >> 4;
    int m    = lane & 15;
    const float* p = base + (size_t)m * ld + 8 * half;
    v16h r;
#pragma unroll
    for (int i = 0; i < 8; ++i) {
        r[i]     = (_Float16)p[i];
        r[i + 8] = (_Float16)p[16 + i];
    }
    return r;
}

// B 32x16 (KxN) f16, stored as M[n][k] row-major (i.e. B = M^T):
// lane-half 0 holds K in {0..7, 16..23}, half 1 {8..15, 24..31}, column n = lane&15.
// Works for global or LDS source (addrspace inferred).
__device__ __forceinline__ v16h load_b_f16(const _Float16* base, int ld) {
    int lane = threadIdx.x & 31;
    int half = lane >> 4;
    int n    = lane & 15;
    const _Float16* p = base + (size_t)n * ld + 8 * half;
    v8h lo = *(const v8h*)p;
    v8h hi = *(const v8h*)(p + 16);
    v16h r;
#pragma unroll
    for (int i = 0; i < 8; ++i) { r[i] = lo[i]; r[i + 8] = hi[i]; }
    return r;
}

__device__ __forceinline__ v8f wmma_f16(v16h a, v16h b, v8f c) {
    return __builtin_amdgcn_wmma_f32_16x16x32_f16(false, a, false, b, (short)0, c,
                                                  false, false);
}

// C/D 16x16 f32: VGPR r -> row (r + 8*half), column = lane&15
__device__ __forceinline__ void store_c_f32(float* __restrict__ base, int ld, v8f c,
                                            float bias_n) {
    int lane = threadIdx.x & 31;
    int half = lane >> 4;
    int n    = lane & 15;
#pragma unroll
    for (int r = 0; r < 8; ++r)
        base[(size_t)(r + 8 * half) * ld + n] = c[r] + bias_n;
}

__device__ __forceinline__ void store_c_f16(_Float16* __restrict__ base, int ld, v8f c,
                                            float bias_n) {
    int lane = threadIdx.x & 31;
    int half = lane >> 4;
    int n    = lane & 15;
#pragma unroll
    for (int r = 0; r < 8; ++r)
        base[(size_t)(r + 8 * half) * ld + n] = (_Float16)(c[r] + bias_n);
}

// ---------------- Tensor Data Mover: stage 64 x AV_CHUNK f16 tile into LDS --
// Source: row-major (DH rows, row stride SK elements), elem = 2 bytes.
// D# built per CDNA5 ISA (2D tile, groups 2/3 zero).
// This toolchain exposes the 6-arg builtin:
//   (uint32x4 g0, int32x8 g1, int32x4 g2, int32x4 g3, int32x8 g4, i32 cpol)
__device__ __forceinline__ void tdm_load_vchunk(const _Float16* gsrc, unsigned lds_off) {
    unsigned long long ga = (unsigned long long)(size_t)gsrc;
    v4u g0;
    g0[0] = 1u;                                   // count=1, user descriptor
    g0[1] = lds_off;                              // lds_addr (bytes)
    g0[2] = (unsigned)(ga & 0xFFFFFFFFu);         // global_addr[31:0]
    g0[3] = (unsigned)((ga >> 32) & 0x01FFFFFFu)  // global_addr[56:32]
          | (2u << 30);                           // type = 2 ("image")
    v8i g1;
    g1[0] = (int)(1u << 16);                      // data_size = 1 -> 2 bytes
    g1[1] = (int)((unsigned)SK << 16);            // tensor_dim0[15:0]
    g1[2] = (int)(((unsigned)SK >> 16) | ((unsigned)DH << 16)); // dim0 hi | dim1 lo
    g1[3] = (int)((unsigned)AV_CHUNK << 16);      // dim1 hi=0 | tile_dim0
    g1[4] = (int)(unsigned)DH;                    // tile_dim1 | tile_dim2=0
    g1[5] = (int)(unsigned)SK;                    // tensor_dim0_stride[31:0]
    g1[6] = 0;                                    // stride0 hi | stride1 lo
    g1[7] = 0;                                    // stride1 hi
    v4i z4 = {0, 0, 0, 0};
    v8i z8 = {0, 0, 0, 0, 0, 0, 0, 0};
    __builtin_amdgcn_tensor_load_to_lds(g0, g1, z4, z4, z8, 0);
}

// ---------------- kernel 1: f32 -> f16 convert ------------------------------
__global__ __launch_bounds__(256) void mha_cvt_f16(const float* __restrict__ src,
                                                   _Float16* __restrict__ dst, int n) {
    int i = blockIdx.x * 256 + threadIdx.x;
    if (i < n) dst[i] = (_Float16)src[i];
}

// ---------------- kernel 2: C = A(M,1024) x W(1024,1024)^T + bias -----------
// block = 256 threads (8 waves), wave grid 4(M) x 2(N): block tile 64 x 128.
// Register double-buffered k-loop: loads for k+32 in flight under current WMMAs.
template <bool STOREF16>
__global__ __launch_bounds__(256) void mha_gemm_xwt(const _Float16* __restrict__ A,
                                                    const _Float16* __restrict__ W,
                                                    const float* __restrict__ bias,
                                                    void* __restrict__ Cout) {
    const int w     = threadIdx.x >> 5;
    const int lane  = threadIdx.x & 31;
    const int row0  = blockIdx.y * 64 + (w >> 1) * 16;
    const int col0  = blockIdx.x * 128 + (w & 1) * 64;

    const _Float16* Abase = A + (size_t)row0 * EM;
    v8f acc[4] = {v8f{0}, v8f{0}, v8f{0}, v8f{0}};

    v16h a_cur = load_a_f16(Abase, EM);
    v16h b_cur[4];
#pragma unroll
    for (int t = 0; t < 4; ++t)
        b_cur[t] = load_b_f16(W + (size_t)(col0 + 16 * t) * EM, EM);

#pragma unroll 4
    for (int k0 = 0; k0 < EM; k0 += 32) {
        v16h a_nxt = a_cur;
        v16h b_nxt[4] = {b_cur[0], b_cur[1], b_cur[2], b_cur[3]};
        const int kn = k0 + 32;
        if (kn < EM) {
            a_nxt = load_a_f16(Abase + kn, EM);
#pragma unroll
            for (int t = 0; t < 4; ++t)
                b_nxt[t] = load_b_f16(W + (size_t)(col0 + 16 * t) * EM + kn, EM);
        }
#pragma unroll
        for (int t = 0; t < 4; ++t)
            acc[t] = wmma_f16(a_cur, b_cur[t], acc[t]);
        a_cur = a_nxt;
#pragma unroll
        for (int t = 0; t < 4; ++t) b_cur[t] = b_nxt[t];
    }

    const int n = lane & 15;
#pragma unroll
    for (int t = 0; t < 4; ++t) {
        float bn = bias[col0 + 16 * t + n];
        if (STOREF16)
            store_c_f16((_Float16*)Cout + (size_t)row0 * EM + col0 + 16 * t, EM, acc[t], bn);
        else
            store_c_f32((float*)Cout + (size_t)row0 * EM + col0 + 16 * t, EM, acc[t], bn);
    }
}

// ---------------- kernel 3: V head-transpose --------------------------------
// Vt[nb][h][d][s] = Vproj[nb*SK+s][h*64+d]
__global__ __launch_bounds__(256) void mha_vtrans(const _Float16* __restrict__ Vp,
                                                  _Float16* __restrict__ Vt) {
    int idx = blockIdx.x * 256 + threadIdx.x;              // over NB*HH*DH*SK = 4M
    int s  = idx & (SK - 1);
    int d  = (idx >> 11) & (DH - 1);
    int h  = (idx >> 17) & (HH - 1);
    int nb = idx >> 21;
    Vt[idx] = Vp[((size_t)(nb * SK + s)) * EM + h * DH + d];
}

// ---------------- kernel 4: scores = (Q @ K^T) / 64 -> attn region ----------
// grid: x = S/256, y = L/32, z = nb*H+h. wave grid 2(L) x 4(S).
__global__ __launch_bounds__(256) void mha_scores(const _Float16* __restrict__ Q16,
                                                  const _Float16* __restrict__ K16,
                                                  float* __restrict__ attn) {
    const int w    = threadIdx.x >> 5;
    const int nh   = blockIdx.z;
    const int nb   = nh / HH;
    const int h    = nh % HH;
    const int l0   = blockIdx.y * 32 + (w >> 2) * 16;
    const int s0   = blockIdx.x * 256 + (w & 3) * 64;

    const _Float16* qbase = Q16 + ((size_t)(nb * LQ + l0)) * EM + h * DH;
    const _Float16* kbase = K16 + ((size_t)nb * SK) * EM + h * DH;

    v8f acc[4] = {v8f{0}, v8f{0}, v8f{0}, v8f{0}};
#pragma unroll
    for (int k0 = 0; k0 < DH; k0 += 32) {
        v16h a = load_a_f16(qbase + k0, EM);
#pragma unroll
        for (int t = 0; t < 4; ++t) {
            v16h b = load_b_f16(kbase + (size_t)(s0 + 16 * t) * EM + k0, EM);
            acc[t] = wmma_f16(a, b, acc[t]);
        }
    }

    const float scale = 1.0f / 64.0f;  // reference double-scales: 1/sqrt(D) twice
    float* obase = attn + ((size_t)nh * LQ + l0) * SK + s0;
#pragma unroll
    for (int t = 0; t < 4; ++t) {
        v8f s = acc[t];
#pragma unroll
        for (int r = 0; r < 8; ++r) s[r] *= scale;
        store_c_f32(obase + 16 * t, SK, s, 0.0f);
    }
}

// ---------------- kernel 5: in-place row softmax (row = 2048 f32) -----------
__global__ __launch_bounds__(256) void mha_softmax(float* __restrict__ attn) {
    __shared__ float red[256];
    float* row = attn + (size_t)blockIdx.x * SK;
    const int t = threadIdx.x;

    float vals[8];
    float m = -1e30f;
#pragma unroll
    for (int j = 0; j < 8; ++j) {
        vals[j] = row[t + 256 * j];
        m = fmaxf(m, vals[j]);
    }
    red[t] = m;
    __syncthreads();
    for (int off = 128; off > 0; off >>= 1) {
        if (t < off) red[t] = fmaxf(red[t], red[t + off]);
        __syncthreads();
    }
    const float rowmax = red[0];
    __syncthreads();

    float s = 0.0f;
#pragma unroll
    for (int j = 0; j < 8; ++j) {
        vals[j] = __expf(vals[j] - rowmax);
        s += vals[j];
    }
    red[t] = s;
    __syncthreads();
    for (int off = 128; off > 0; off >>= 1) {
        if (t < off) red[t] += red[t + off];
        __syncthreads();
    }
    const float inv = 1.0f / red[0];
#pragma unroll
    for (int j = 0; j < 8; ++j) row[t + 256 * j] = vals[j] * inv;
}

// ---------------- kernel 6: O = attn @ V (per head), TDM-staged V -----------
// grid: x = L/128, z = nb*H+h. 8 waves, each one 16-row L tile covering D=64.
// V head slice (64 x 2048 f16) streamed through LDS in 4 chunks of 64 x 512,
// double-buffered: TDM of chunk c+1 overlaps WMMA on chunk c.
__global__ __launch_bounds__(256) void mha_av(const float* __restrict__ attn,
                                              const _Float16* __restrict__ Vt,
                                              _Float16* __restrict__ O16) {
    extern __shared__ _Float16 smem[];   // 2 * 64 * 512 f16 = 128 KB
    const int w  = threadIdx.x >> 5;
    const int nh = blockIdx.z;
    const int nb = nh / HH;
    const int h  = nh % HH;
    const int l0 = blockIdx.x * 128 + w * 16;

    const float*    abase = attn + ((size_t)nh * LQ + l0) * SK;
    const _Float16* vbase = Vt + (size_t)nh * DH * SK;
    const unsigned  lds_base = (unsigned)(size_t)(void*)smem;  // LDS byte offset

    v8f acc[4] = {v8f{0}, v8f{0}, v8f{0}, v8f{0}};

    if (w == 0) tdm_load_vchunk(vbase, lds_base);              // chunk 0 -> buf 0

    for (int c = 0; c < SK / AV_CHUNK; ++c) {
        __syncthreads();  // all waves done reading buf[(c+1)&1] (chunk c-1)
        if (w == 0) {
            if (c + 1 < SK / AV_CHUNK) {
                tdm_load_vchunk(vbase + (size_t)(c + 1) * AV_CHUNK,
                                lds_base + (unsigned)((c + 1) & 1) * AV_CHUNK_BYTES);
                __builtin_amdgcn_s_wait_tensorcnt(1);  // chunk c complete (in-order)
            } else {
                __builtin_amdgcn_s_wait_tensorcnt(0);
            }
        }
        __syncthreads();  // chunk c visible to all waves

        const _Float16* vb = smem + (size_t)(c & 1) * (DH * AV_CHUNK);
#pragma unroll 4
        for (int kk = 0; kk < AV_CHUNK; kk += 32) {
            v16h a = load_a_f32cvt(abase + c * AV_CHUNK + kk, SK);
#pragma unroll
            for (int t = 0; t < 4; ++t) {
                v16h b = load_b_f16(vb + (size_t)(16 * t) * AV_CHUNK + kk, AV_CHUNK);
                acc[t] = wmma_f16(a, b, acc[t]);
            }
        }
    }

    _Float16* obase = O16 + ((size_t)(nb * LQ + l0)) * EM + h * DH;
#pragma unroll
    for (int t = 0; t < 4; ++t)
        store_c_f16(obase + 16 * t, EM, acc[t], 0.0f);
}

// ---------------- host launcher ---------------------------------------------
extern "C" void kernel_launch(void* const* d_in, const int* in_sizes, int n_in,
                              void* d_out, int out_size, void* d_ws, size_t ws_size,
                              hipStream_t stream) {
    (void)in_sizes; (void)n_in; (void)out_size; (void)ws_size;

    const float* q_in = (const float*)d_in[0];
    const float* k_in = (const float*)d_in[1];
    const float* v_in = (const float*)d_in[2];
    const float* Wq   = (const float*)d_in[3];
    const float* Wk   = (const float*)d_in[4];
    const float* Wv   = (const float*)d_in[5];
    const float* Wo   = (const float*)d_in[6];
    const float* bq   = (const float*)d_in[7];
    const float* bk   = (const float*)d_in[8];
    const float* bv   = (const float*)d_in[9];
    const float* bo   = (const float*)d_in[10];

    float* out_f32  = (float*)d_out;                       // (N, L, E)
    float* attn_f32 = out_f32 + (size_t)NT * EM;           // (N, H, L, S)

    // workspace layout (f16 elements)
    _Float16* w16 = (_Float16*)d_ws;
    const size_t XSZ = (size_t)NT * EM;   // 4M elements
    const size_t WSZ = (size_t)EM * EM;   // 1M elements
    _Float16* q_in16 = w16;               // reused later as O16
    _Float16* k_in16 = w16 + XSZ;
    _Float16* v_in16 = w16 + 2 * XSZ;     // reused later as Vt
    _Float16* Wq16   = w16 + 3 * XSZ;
    _Float16* Wk16   = Wq16 + WSZ;
    _Float16* Wv16   = Wk16 + WSZ;
    _Float16* Wo16   = Wv16 + WSZ;
    _Float16* Q16    = Wo16 + WSZ;
    _Float16* K16    = Q16 + XSZ;
    _Float16* Vp16   = K16 + XSZ;
    _Float16* O16    = q_in16;            // alias: q_in16 dead after projections
    _Float16* Vt16   = v_in16;            // alias: v_in16 dead after V projection

    dim3 blk(256);

    // 1) convert activations + weights to f16
    mha_cvt_f16<<<dim3(XSZ / 256), blk, 0, stream>>>(q_in, q_in16, (int)XSZ);
    mha_cvt_f16<<<dim3(XSZ / 256), blk, 0, stream>>>(k_in, k_in16, (int)XSZ);
    mha_cvt_f16<<<dim3(XSZ / 256), blk, 0, stream>>>(v_in, v_in16, (int)XSZ);
    mha_cvt_f16<<<dim3(WSZ / 256), blk, 0, stream>>>(Wq, Wq16, (int)WSZ);
    mha_cvt_f16<<<dim3(WSZ / 256), blk, 0, stream>>>(Wk, Wk16, (int)WSZ);
    mha_cvt_f16<<<dim3(WSZ / 256), blk, 0, stream>>>(Wv, Wv16, (int)WSZ);
    mha_cvt_f16<<<dim3(WSZ / 256), blk, 0, stream>>>(Wo, Wo16, (int)WSZ);

    // 2) projections (f16 out): grid (EM/128, NT/64)
    dim3 ggrid(EM / 128, NT / 64);
    mha_gemm_xwt<true><<<ggrid, blk, 0, stream>>>(q_in16, Wq16, bq, (void*)Q16);
    mha_gemm_xwt<true><<<ggrid, blk, 0, stream>>>(k_in16, Wk16, bk, (void*)K16);
    mha_gemm_xwt<true><<<ggrid, blk, 0, stream>>>(v_in16, Wv16, bv, (void*)Vp16);

    // 3) V head-transpose -> (N,H,D,S)
    mha_vtrans<<<dim3(XSZ / 256), blk, 0, stream>>>(Vp16, Vt16);

    // 4) scores -> attn region of d_out (raw, scaled by 1/64)
    mha_scores<<<dim3(SK / 256, LQ / 32, NB * HH), blk, 0, stream>>>(Q16, K16, attn_f32);

    // 5) in-place softmax over S
    mha_softmax<<<dim3(NB * HH * LQ), blk, 0, stream>>>(attn_f32);

    // 6) O = attn @ V (f16), V staged through LDS by the Tensor Data Mover
    mha_av<<<dim3(LQ / 128, 1, NB * HH), blk, 2 * AV_CHUNK_BYTES, stream>>>(
        attn_f32, Vt16, O16);

    // 7) output = O @ Wo^T + bo (f32 into d_out)
    mha_gemm_xwt<false><<<ggrid, blk, 0, stream>>>(O16, Wo16, bo, (void*)out_f32);
}
